// HierarchicalAttention_66082366816989
// MI455X (gfx1250) — matrix operand order repaired and checked by reading, not compiled, not run on hardware
//
#include <hip/hip_runtime.h>
#include <hip/hip_bf16.h>

#define DEVICE __device__ __forceinline__

typedef __attribute__((ext_vector_type(16))) __bf16 v16bf;
typedef __attribute__((ext_vector_type(8)))  __bf16 v8bf;
typedef __attribute__((ext_vector_type(8)))  float  v8f;
typedef __attribute__((ext_vector_type(4)))  int    v4i_t;

static_assert(sizeof(__bf16) == 2, "bf16 must be 2 bytes");

// ---------------------------------------------------------------------------
// WMMA helper (CDNA5 wave32, V_WMMA_F32_16X16X32_BF16)
// ---------------------------------------------------------------------------
DEVICE v8f wmma_bf16(v16bf a, v16bf b, v8f c) {
  return __builtin_amdgcn_wmma_f32_16x16x32_bf16(
      false, a, false, b, (short)0, c, false, false);
}

// Async global->LDS 16-byte copy (gfx1250 GLOBAL_LOAD_ASYNC_TO_LDS_B128,
// tracked by ASYNCcnt). Falls back to a regular load+store pair if the
// toolchain doesn't expose the builtin.
#if defined(__HIP_DEVICE_COMPILE__) && __has_builtin(__builtin_amdgcn_global_load_async_to_lds_b128) && __has_builtin(__builtin_amdgcn_s_wait_asynccnt)
#define HAVE_ASYNC_LDS 1
#else
#define HAVE_ASYNC_LDS 0
#endif

DEVICE void copy16_to_lds(const __bf16* g, __bf16* l) {
#if HAVE_ASYNC_LDS
  __builtin_amdgcn_global_load_async_to_lds_b128(
      (__attribute__((address_space(1))) v4i_t*)g,
      (__attribute__((address_space(3))) v4i_t*)l, 0, 0);
#else
  *(v8bf*)l = *(const v8bf*)g;
#endif
}

DEVICE void async_lds_wait() {
#if HAVE_ASYNC_LDS
  __builtin_amdgcn_s_wait_asynccnt(0);
#endif
}

// ---------------------------------------------------------------------------
// Fragment loaders (k = 16*half + s in A and B alike; per-lane data is contiguous so the
// compiler lowers each to 2x (ds|global)_load_b128).
// ---------------------------------------------------------------------------
DEVICE v16bf load_a_frag(const __bf16* A, int pitch, int m0, int k0, int lane) {
  int half = lane >> 4, mn = lane & 15;
  const __bf16* row = A + (size_t)(m0 + mn) * pitch + k0;
  v16bf a;
#pragma unroll
  for (int v = 0; v < 8; ++v) {
    int kb = half * 16 + v * 2;
    a[2 * v]     = row[kb];
    a[2 * v + 1] = row[kb + 1];
  }
  return a;
}

// B fragment from a K-contiguous source rows[n][k] (transposed LDS tile or
// [N][K]-major global matrix). Lane's 16 K elems are contiguous.
DEVICE v16bf load_bT_frag(const __bf16* rows, int pitch, int n0, int k0, int lane) {
  int half = lane >> 4, nn = lane & 15;
  const __bf16* row = rows + (size_t)(n0 + nn) * pitch + k0 + half * 16;
  v16bf b;
#pragma unroll
  for (int i = 0; i < 16; ++i) b[i] = row[i];
  return b;
}

// ---------------------------------------------------------------------------
// Kernel 1: projection GEMM  C_bf16[M,N] = A_f32[M,K] * W_f32[K,N] + bias
// 256 thr (8 waves). Tile 128 x BN, K-step 32. W staged transposed.
// Wave w owns rows w*16..w*16+15 and all BN/16 col tiles (A-frag reuse).
// TRANS_OUT: write C as Vt[bt][n][j] with m = bt*256 + j (for the V matrix),
// so the consumer's B tiles are [n][k]-contiguous in global memory. The 8
// accumulator rows of a lane are 8 consecutive j -> one 16B vector store.
// ---------------------------------------------------------------------------
template <int BN, bool TRANS_OUT>
__global__ __launch_bounds__(256) void proj_gemm_kernel(
    const float* __restrict__ A, const float* __restrict__ W,
    const float* __restrict__ bias, __bf16* __restrict__ C,
    int M, int N, int K) {
  constexpr int NCT = BN / 16;
  __shared__ __bf16 lA[128 * 32];
  __shared__ __bf16 lBt[BN * 32];
  const int m_blk = blockIdx.x * 128;
  const int n_blk = blockIdx.y * BN;
  const int tid = threadIdx.x, wave = tid >> 5, lane = tid & 31;
  v8f acc[NCT] = {};

  for (int k0 = 0; k0 < K; k0 += 32) {
    // stage A 128x32 (f32 -> bf16), float4 reads
#pragma unroll
    for (int it = 0; it < 4; ++it) {
      int idx = tid + it * 256;             // 1024 float4 chunks
      int r = idx >> 3, c4 = (idx & 7) * 4; // 8 chunks per row
      float4 f = *(const float4*)(A + (size_t)(m_blk + r) * K + k0 + c4);
      __bf16* dst = lA + r * 32 + c4;
      dst[0] = (__bf16)f.x; dst[1] = (__bf16)f.y;
      dst[2] = (__bf16)f.z; dst[3] = (__bf16)f.w;
    }
    // stage W 32xBN transposed into lBt[n][k]
#pragma unroll
    for (int it = 0; it < (32 * BN / 4) / 256; ++it) {
      int idx = tid + it * 256;
      int r = idx / (BN / 4), c = (idx % (BN / 4)) * 4;
      float4 f = *(const float4*)(W + (size_t)(k0 + r) * N + n_blk + c);
      lBt[(c + 0) * 32 + r] = (__bf16)f.x;
      lBt[(c + 1) * 32 + r] = (__bf16)f.y;
      lBt[(c + 2) * 32 + r] = (__bf16)f.z;
      lBt[(c + 3) * 32 + r] = (__bf16)f.w;
    }
    __syncthreads();
    v16bf af = load_a_frag(lA, 32, wave * 16, 0, lane);
#pragma unroll
    for (int ct = 0; ct < NCT; ++ct)
      acc[ct] = wmma_bf16(af, load_bT_frag(lBt, 32, ct * 16, 0, lane), acc[ct]);
    __syncthreads();
  }

  const int half = lane >> 4, nn = lane & 15;
  const int m0 = m_blk + wave * 16 + half * 8;   // 8 consecutive rows m0..m0+7
#pragma unroll
  for (int ct = 0; ct < NCT; ++ct) {
    int n = n_blk + ct * 16 + nn;
    float bn = bias[n];
    if (TRANS_OUT) {
      // rows m0..m0+7 are 8 consecutive j in Vt[bt][n][j]
      v8bf pk;
#pragma unroll
      for (int v = 0; v < 8; ++v) pk[v] = (__bf16)(acc[ct][v] + bn);
      *(v8bf*)(C + ((size_t)(m0 >> 8) * N + n) * 256 + (m0 & 255)) = pk;
    } else {
#pragma unroll
      for (int v = 0; v < 8; ++v)
        C[(size_t)(m0 + v) * N + n] = (__bf16)(acc[ct][v] + bn);
    }
  }
}

// ---------------------------------------------------------------------------
// Kernel 2: per-(b,t) attention scores + within-sequence softmax.
// S[i0..i0+15][0..255] = Q[b] * K[b,t]^T  (contraction kd=64).
// K is stored [j][kd] = [N][K]-major -> B fragments straight from global.
// Writes P (softmax over j, bf16) and raw row max (for level 2).
// ---------------------------------------------------------------------------
__global__ __launch_bounds__(256) void attn_softmax_kernel(
    const __bf16* __restrict__ Q,    // [b,512,64]
    const __bf16* __restrict__ Km,   // [b,t,256,64]
    __bf16* __restrict__ P,          // [b,t,512,256]
    float* __restrict__ Mrow,        // [b,t,512]
    float scale) {
  const int i0 = blockIdx.x * 16;
  const int t = blockIdx.y, b = blockIdx.z;
  const __bf16* Qb  = Q  + (size_t)b * 512 * 64;
  const __bf16* Kbt = Km + (size_t)(b * 32 + t) * 256 * 64;
  __bf16* Pbt = P + (size_t)(b * 32 + t) * 512 * 256;

  __shared__ float S[16 * 256];
  __shared__ float red[16 * 16];
  __shared__ float rmax[16], rsum[16];

  const int tid = threadIdx.x, wave = tid >> 5, lane = tid & 31;

  // Each wave owns 32 columns: j = wave*32 .. wave*32+31 (2 WMMA tiles).
  v8f acc0 = {}, acc1 = {};
  {
    v16bf a0 = load_a_frag(Qb, 64, i0, 0, lane);
    v16bf a1 = load_a_frag(Qb, 64, i0, 32, lane);
    int n0 = wave * 32;
    acc0 = wmma_bf16(a0, load_bT_frag(Kbt, 64, n0,      0,  lane), acc0);
    acc0 = wmma_bf16(a1, load_bT_frag(Kbt, 64, n0,      32, lane), acc0);
    acc1 = wmma_bf16(a0, load_bT_frag(Kbt, 64, n0 + 16, 0,  lane), acc1);
    acc1 = wmma_bf16(a1, load_bT_frag(Kbt, 64, n0 + 16, 32, lane), acc1);
  }
  const int half = lane >> 4, nn = lane & 15;
#pragma unroll
  for (int v = 0; v < 8; ++v) {
    int r = half * 8 + v;
    S[r * 256 + wave * 32 + nn]      = acc0[v];
    S[r * 256 + wave * 32 + 16 + nn] = acc1[v];
  }
  __syncthreads();

  // Softmax over 256 cols per row; 16 threads per row, each owning a
  // contiguous 16-column block (vectorizable LDS/global accesses).
  const int r = tid >> 4, c = tid & 15;
  float* Sr = S + r * 256 + c * 16;
  float mx = -3.0e38f;
#pragma unroll
  for (int q = 0; q < 16; ++q) mx = fmaxf(mx, Sr[q]);
  red[r * 16 + c] = mx;
  __syncthreads();
  if (c == 0) {
    float m2 = red[r * 16];
#pragma unroll
    for (int q = 1; q < 16; ++q) m2 = fmaxf(m2, red[r * 16 + q]);
    rmax[r] = m2;
    Mrow[(size_t)(b * 32 + t) * 512 + i0 + r] = m2;   // raw (unscaled) max
  }
  __syncthreads();
  const float rowmax = rmax[r];
  float sum = 0.f;
#pragma unroll
  for (int q = 0; q < 16; ++q) {
    float e = __expf(scale * (Sr[q] - rowmax));
    Sr[q] = e;
    sum += e;
  }
  red[r * 16 + c] = sum;
  __syncthreads();
  if (c == 0) {
    float s2 = 0.f;
#pragma unroll
    for (int q = 0; q < 16; ++q) s2 += red[r * 16 + q];
    rsum[r] = s2;
  }
  __syncthreads();
  const float inv = 1.0f / rsum[r];
  v8bf p0, p1;
#pragma unroll
  for (int q = 0; q < 8; ++q) {
    p0[q] = (__bf16)(Sr[q] * inv);
    p1[q] = (__bf16)(Sr[q + 8] * inv);
  }
  __bf16* prow = Pbt + (size_t)(i0 + r) * 256 + c * 16;
  *(v8bf*)(prow)     = p0;
  *(v8bf*)(prow + 8) = p1;
}

// ---------------------------------------------------------------------------
// Kernel 3: level-2 softmax over t (32 values per (b,i)).
// ---------------------------------------------------------------------------
__global__ __launch_bounds__(256) void seq_softmax_kernel(
    const float* __restrict__ Mrow,  // [b,t,512]
    float* __restrict__ W2,          // [b,t,512]
    float scale) {
  int idx = blockIdx.x * blockDim.x + threadIdx.x;
  if (idx >= 4 * 512) return;
  int b = idx >> 9, i = idx & 511;
  const float* m = Mrow + (size_t)b * 32 * 512 + i;
  float mx = -3.0e38f;
#pragma unroll
  for (int t = 0; t < 32; ++t) mx = fmaxf(mx, m[t * 512]);
  float sum = 0.f;
#pragma unroll
  for (int t = 0; t < 32; ++t) sum += __expf(scale * (m[t * 512] - mx));
  float inv = 1.0f / sum;
  float* w = W2 + (size_t)b * 32 * 512 + i;
#pragma unroll
  for (int t = 0; t < 32; ++t) w[t * 512] = __expf(scale * (m[t * 512] - mx)) * inv;
}

// ---------------------------------------------------------------------------
// Kernel 4: fused level-2 retrieval as one GEMM per batch:
//   out[b] (512x512) = (W2 ⊙ P)[b] (512 x 8192) * Vt[b]^T (8192 x 512)
// K index kk = t*256 + j. Tile 128x64, K-step 32. W2 row-scaling applied
// during A staging (VALU). V was produced transposed (Vt[bt][n][j]) so the
// B tile is a straight [n][k]-contiguous copy: async global->LDS B128.
// ---------------------------------------------------------------------------
__global__ __launch_bounds__(256) void final_gemm_kernel(
    const __bf16* __restrict__ P,   // [b,t,512,256]
    const __bf16* __restrict__ Vt,  // [b,t,512(dv),256(j)]
    const float* __restrict__ W2,   // [b,t,512]
    float* __restrict__ out) {      // [b,512,512]
  __shared__ __bf16 lA[128 * 32];   // 8 KB
  __shared__ __bf16 lBt[64 * 32];   // 4 KB, [n][k]
  const int b = blockIdx.z;
  const int m_blk = blockIdx.x * 128, n_blk = blockIdx.y * 64;
  const int tid = threadIdx.x, wave = tid >> 5, lane = tid & 31;
  v8f acc0 = {}, acc1 = {}, acc2 = {}, acc3 = {};

  for (int kk = 0; kk < 8192; kk += 32) {
    const int t = kk >> 8, j0 = kk & 255;      // t constant within a 32-step
    const __bf16* Pbt  = P  + (size_t)(b * 32 + t) * 512 * 256;
    const __bf16* Vtbt = Vt + (size_t)(b * 32 + t) * 512 * 256;
    const float*  w2   = W2 + (size_t)(b * 32 + t) * 512;

    // stage B tile (64 n x 32 k): one async 16B chunk per thread
    {
      int n = tid >> 2, c8 = (tid & 3) * 8;
      copy16_to_lds(Vtbt + (size_t)(n_blk + n) * 256 + j0 + c8,
                    lBt + n * 32 + c8);
    }
    // stage A' = W2 * P  (128x32), 8-bf16 (16B) vector reads
#pragma unroll
    for (int it = 0; it < 2; ++it) {
      int idx = tid + it * 256;             // 512 chunks of 8
      int r = idx >> 2, c8 = (idx & 3) * 8; // 4 chunks per row
      v8bf p = *(const v8bf*)(Pbt + (size_t)(m_blk + r) * 256 + j0 + c8);
      float s = w2[m_blk + r];
      __bf16* dst = lA + r * 32 + c8;
#pragma unroll
      for (int q = 0; q < 8; ++q) dst[q] = (__bf16)((float)p[q] * s);
    }
    async_lds_wait();
    __syncthreads();
    v16bf af = load_a_frag(lA, 32, wave * 16, 0, lane);
    acc0 = wmma_bf16(af, load_bT_frag(lBt, 32,  0, 0, lane), acc0);
    acc1 = wmma_bf16(af, load_bT_frag(lBt, 32, 16, 0, lane), acc1);
    acc2 = wmma_bf16(af, load_bT_frag(lBt, 32, 32, 0, lane), acc2);
    acc3 = wmma_bf16(af, load_bT_frag(lBt, 32, 48, 0, lane), acc3);
    __syncthreads();
  }

  const int half = lane >> 4, nn = lane & 15;
  float* ob = out + (size_t)b * 512 * 512;
  const v8f accs[4] = {acc0, acc1, acc2, acc3};
#pragma unroll
  for (int ct = 0; ct < 4; ++ct) {
#pragma unroll
    for (int v = 0; v < 8; ++v) {
      int m = m_blk + wave * 16 + half * 8 + v;
      int n = n_blk + ct * 16 + nn;
      ob[(size_t)m * 512 + n] = accs[ct][v];
    }
  }
}

// ---------------------------------------------------------------------------
// Host launcher
// ---------------------------------------------------------------------------
extern "C" void kernel_launch(void* const* d_in, const int* in_sizes, int n_in,
                              void* d_out, int out_size, void* d_ws, size_t ws_size,
                              hipStream_t stream) {
  (void)in_sizes; (void)n_in; (void)out_size; (void)ws_size;
  const float* input_seq = (const float*)d_in[0];  // [4,512,512]
  const float* memory_x  = (const float*)d_in[1];  // [4,32,256,512]
  const float* memory_y  = (const float*)d_in[2];  // [4,32,256,512]
  const float* Wq = (const float*)d_in[3];         // [512,64]
  const float* bq = (const float*)d_in[4];         // [64]
  const float* Wk = (const float*)d_in[5];         // [512,64]
  const float* bk = (const float*)d_in[6];         // [64]
  const float* Wv = (const float*)d_in[7];         // [512,512]
  const float* bv = (const float*)d_in[8];         // [512]
  float* out = (float*)d_out;                      // [4,512,512]

  const float SCALE = 0.125f;  // 1/sqrt(64)

  // Workspace layout (bf16 activations + fp32 stats), ~72 MB total.
  __bf16* Qbf = (__bf16*)d_ws;                        // 2048*64
  __bf16* Kbf = Qbf + (size_t)2048 * 64;              // 32768*64
  __bf16* Vtb = Kbf + (size_t)32768 * 64;             // [b,t,512,256] transposed V
  __bf16* Pbf = Vtb + (size_t)128 * 512 * 256;        // 4*32*512*256
  float*  Mrw = (float*)(Pbf + (size_t)4 * 32 * 512 * 256);  // 65536
  float*  W2  = Mrw + (size_t)65536;                  // 65536

  // 1) Projections (fp32 in, bf16 out, f32 WMMA accumulate)
  proj_gemm_kernel<64, false><<<dim3(2048 / 128, 1),  256, 0, stream>>>(
      input_seq, Wq, bq, Qbf, 2048, 64, 512);
  proj_gemm_kernel<64, false><<<dim3(32768 / 128, 1), 256, 0, stream>>>(
      memory_x, Wk, bk, Kbf, 32768, 64, 512);
  proj_gemm_kernel<128, true><<<dim3(32768 / 128, 4), 256, 0, stream>>>(
      memory_y, Wv, bv, Vtb, 32768, 512, 512);

  // 2) Scores + within-sequence softmax (+ raw row max for level 2)
  attn_softmax_kernel<<<dim3(512 / 16, 32, 4), 256, 0, stream>>>(Qbf, Kbf, Pbf, Mrw, SCALE);

  // 3) Over-sequences softmax
  seq_softmax_kernel<<<dim3(8), 256, 0, stream>>>(Mrw, W2, SCALE);

  // 4) Fused retrieval: out[b] = (W2 ⊙ P) * Vt^T, K = 8192
  final_gemm_kernel<<<dim3(4, 8, 4), 256, 0, stream>>>(Pbf, Vtb, W2, out);
}
